// GatedCrossAttention_78494822302252
// MI455X (gfx1250) — matrix operand order repaired and checked by reading, not compiled, not run on hardware
//
#include <hip/hip_runtime.h>
#include <hip/hip_bf16.h>
#include <cstdint>

// Problem constants (from reference): S=2048, B=4, D=1024, Z=128, MP=2048
#define S_LEN 2048
#define BATCH 4
#define DIM   1024
#define ZDIM  128
#define MPOS  2048
#define MROWS (S_LEN * BATCH)   // 8192 fused (s,b) rows

// Block tiling: 256 threads = 8 waves in a 4(M) x 2(N) grid, each wave owns a
// 64x64 tile (4x4 WMMA fragments) -> block tile 256(M) x 128(N).
#define BLK_M 256
#define BLK_N 128

typedef __bf16 bf16_t;
typedef bf16_t v16bf __attribute__((ext_vector_type(16)));
typedef float  v8f   __attribute__((ext_vector_type(8)));

// ---------- scalar helpers ----------
__device__ __forceinline__ unsigned short f2bf(float f) {
    union { float f; unsigned int u; } c; c.f = f;
    unsigned int r = (c.u + 0x7FFFu + ((c.u >> 16) & 1u)) >> 16;  // RNE
    return (unsigned short)r;
}
__device__ __forceinline__ float bf2f(unsigned short h) {
    union { float f; unsigned int u; } c; c.u = ((unsigned int)h) << 16;
    return c.f;
}
__device__ __forceinline__ float sigmoidf_(float x) { return 1.0f / (1.0f + __expf(-x)); }

// ---------- WMMA fragment loads (gfx1250 bf16 16x16x32 layouts) ----------
// A (16x32, 16-bit): lane l: m = l&15; lane-group g = l>>4.
//   elements 0..7  (VGPR0-3) = K [8g   .. 8g+7]
//   elements 8..15 (VGPR4-7) = K [16+8g .. 16+8g+7]
__device__ __forceinline__ v16bf load_frag_a(const unsigned short* __restrict__ A,
                                             int lda, int m0, int k0) {
    int l = threadIdx.x & 31;
    int g = l >> 4, mi = l & 15;
    const unsigned short* p = A + (size_t)(m0 + mi) * lda + k0;
    union { uint4 q[2]; v16bf v; } u;
    u.q[0] = *(const uint4*)(p + 8 * g);
    u.q[1] = *(const uint4*)(p + 16 + 8 * g);
    return u.v;
}
// B (32x16, 16-bit), stored row-major (N,K): lane l: n = l&15; g = l>>4.
//   elements 0..7 = K [8g .. 8g+7], elements 8..15 = K [16+8g .. 16+8g+7]
__device__ __forceinline__ v16bf load_frag_b(const unsigned short* __restrict__ B,
                                             int ldb, int n0, int k0) {
    int l = threadIdx.x & 31;
    int g = l >> 4, ni = l & 15;
    const unsigned short* p = B + (size_t)(n0 + ni) * ldb + k0;
    union { uint4 q[2]; v16bf v; } u;
    u.q[0] = *(const uint4*)(p + 8 * g);
    u.q[1] = *(const uint4*)(p + 16 + 8 * g);
    return u.v;
}

// ---------- WMMA GEMM core: wave computes MW*16 x NW*16 of C = A(MxK) * B(NxK)^T ----------
template <int MW, int NW>
__device__ __forceinline__ void wmma_gemm(const unsigned short* __restrict__ A, int lda,
                                          const unsigned short* __restrict__ B, int ldb,
                                          int K, int m0w, int n0w, v8f acc[MW][NW]) {
#pragma unroll
    for (int i = 0; i < MW; ++i)
#pragma unroll
        for (int j = 0; j < NW; ++j) acc[i][j] = (v8f){};
    for (int k0 = 0; k0 < K; k0 += 32) {
        v16bf a[MW], b[NW];
#pragma unroll
        for (int i = 0; i < MW; ++i) a[i] = load_frag_a(A, lda, m0w + 16 * i, k0);
#pragma unroll
        for (int j = 0; j < NW; ++j) b[j] = load_frag_b(B, ldb, n0w + 16 * j, k0);
#pragma unroll
        for (int i = 0; i < MW; ++i)
#pragma unroll
            for (int j = 0; j < NW; ++j)
                acc[i][j] = __builtin_amdgcn_wmma_f32_16x16x32_bf16(
                    false, a[i], false, b[j], (short)0, acc[i][j], false, false);
    }
}

// C/D layout (f32 16x16): lane l: n = l&15; element e: m = e + 8*(l>>4)
// Variadic so commas inside the body don't split macro arguments.
#define EPILOGUE_LOOP(...)                                        \
    do {                                                          \
        int l_ = threadIdx.x & 31;                                \
        int g_ = l_ >> 4;                                         \
        int nl_ = l_ & 15;                                        \
        _Pragma("unroll") for (int i = 0; i < 4; ++i)             \
        _Pragma("unroll") for (int j = 0; j < 4; ++j)             \
        _Pragma("unroll") for (int e = 0; e < 8; ++e) {           \
            int m = m0w + 16 * i + 8 * g_ + e;                    \
            int n = n0w + 16 * j + nl_;                           \
            float x = acc[i][j][e];                               \
            __VA_ARGS__                                           \
        }                                                         \
    } while (0)

#define WAVE_TILE_SETUP()                                         \
    int wave = threadIdx.x >> 5;                                  \
    int m0w = blockIdx.x * BLK_M + (wave >> 1) * 64;              \
    int n0w = blockIdx.y * BLK_N + (wave & 1) * 64;               \
    v8f acc[4][4];

// ---------- elementwise kernels ----------
__global__ void cast_kernel(const float* __restrict__ in, unsigned short* __restrict__ out, int n) {
    int i = blockIdx.x * 256 + threadIdx.x;
    if (i < n) out[i] = f2bf(in[i]);
}

__global__ __launch_bounds__(256) void ln_kernel(const float* __restrict__ x,
                                                 const float* __restrict__ gamma,
                                                 const float* __restrict__ beta,
                                                 unsigned short* __restrict__ out) {
    int row = blockIdx.x;
    const float* xr = x + (size_t)row * DIM;
    float s = 0.f, s2 = 0.f;
    for (int i = threadIdx.x; i < DIM; i += 256) { float v = xr[i]; s += v; s2 += v * v; }
    __shared__ float red[256], red2[256];
    red[threadIdx.x] = s; red2[threadIdx.x] = s2;
    __syncthreads();
    for (int st = 128; st > 0; st >>= 1) {
        if (threadIdx.x < st) { red[threadIdx.x] += red[threadIdx.x + st];
                                red2[threadIdx.x] += red2[threadIdx.x + st]; }
        __syncthreads();
    }
    float mu = red[0] * (1.0f / DIM);
    float var = red2[0] * (1.0f / DIM) - mu * mu;
    float rs = rsqrtf(var + 1e-5f);
    for (int i = threadIdx.x; i < DIM; i += 256)
        out[(size_t)row * DIM + i] = f2bf((xr[i] - mu) * rs * gamma[i] + beta[i]);
}

// ---------- GEMM stage kernels ----------
// base = qn @ w_q^T + b_q ; split into u=sigmoid(f32), r=silu(bf16), q(bf16)
__global__ __launch_bounds__(256) void gemm_q_kernel(const unsigned short* __restrict__ qn,
                                                     const unsigned short* __restrict__ wq,
                                                     const float* __restrict__ bq,
                                                     float* __restrict__ u,
                                                     unsigned short* __restrict__ r,
                                                     unsigned short* __restrict__ q) {
    WAVE_TILE_SETUP();
    wmma_gemm<4, 4>(qn, DIM, wq, DIM, DIM, m0w, n0w, acc);
    EPILOGUE_LOOP(
        x += bq[n];
        if (n < DIM)            u[(size_t)m * DIM + n]          = sigmoidf_(x);
        else if (n < 2 * DIM)   r[(size_t)m * DIM + (n - DIM)]  = f2bf(x * sigmoidf_(x));
        else                    q[(size_t)m * ZDIM + (n - 2 * DIM)] = f2bf(x);
    );
}

// k = key @ w_k^T + b_k  (N=128)
__global__ __launch_bounds__(256) void gemm_k_kernel(const unsigned short* __restrict__ keyb,
                                                     const unsigned short* __restrict__ wk,
                                                     const float* __restrict__ bk,
                                                     unsigned short* __restrict__ kout) {
    WAVE_TILE_SETUP();
    wmma_gemm<4, 4>(keyb, DIM, wk, DIM, DIM, m0w, n0w, acc);
    EPILOGUE_LOOP(
        kout[(size_t)m * ZDIM + n] = f2bf(x + bk[n]);
    );
}

// v = silu(key @ w_v^T + b_v), stored TRANSPOSED as v_t[b][d][s] for attn@v B-fragments
__global__ __launch_bounds__(256) void gemm_v_kernel(const unsigned short* __restrict__ keyb,
                                                     const unsigned short* __restrict__ wv,
                                                     const float* __restrict__ bv,
                                                     unsigned short* __restrict__ vt) {
    WAVE_TILE_SETUP();
    wmma_gemm<4, 4>(keyb, DIM, wv, DIM, DIM, m0w, n0w, acc);
    EPILOGUE_LOOP(
        x += bv[n];
        float sv = x * sigmoidf_(x);
        int srow = m >> 2;         // m = s*B + b, B=4
        int bb = m & 3;
        vt[((size_t)bb * DIM + n) * S_LEN + srow] = f2bf(sv);
    );
}

// attn[b,s,c] = relu(q.k/clen + rel_bias)^2   (per-batch, K=Z=128)
__global__ __launch_bounds__(256) void qk_kernel(const unsigned short* __restrict__ qb,
                                                 const unsigned short* __restrict__ kb,
                                                 const float* __restrict__ rel,
                                                 unsigned short* __restrict__ attn) {
    int b = blockIdx.z;
    const unsigned short* A  = qb + b * ZDIM;   // row stride B*Z
    const unsigned short* Bm = kb + b * ZDIM;
    WAVE_TILE_SETUP();
    wmma_gemm<4, 4>(A, BATCH * ZDIM, Bm, BATCH * ZDIM, ZDIM, m0w, n0w, acc);
    const float inv = 1.0f / (float)S_LEN;
    unsigned short* out = attn + (size_t)b * S_LEN * S_LEN;
    EPILOGUE_LOOP(
        float y = x * inv + rel[MPOS - 1 + n - m];
        float t = y > 0.f ? y * y : 0.f;
        out[(size_t)m * S_LEN + n] = f2bf(t);
    );
}

// h = attn @ v ; hr = h * r (bf16) for final projection   (per-batch, K=S)
__global__ __launch_bounds__(256) void av_kernel(const unsigned short* __restrict__ attn,
                                                 const unsigned short* __restrict__ vt,
                                                 const unsigned short* __restrict__ r,
                                                 unsigned short* __restrict__ hr) {
    int b = blockIdx.z;
    const unsigned short* A  = attn + (size_t)b * S_LEN * S_LEN;
    const unsigned short* Bm = vt + (size_t)b * DIM * S_LEN;
    WAVE_TILE_SETUP();
    wmma_gemm<4, 4>(A, S_LEN, Bm, S_LEN, S_LEN, m0w, n0w, acc);
    EPILOGUE_LOOP(
        size_t idx = ((size_t)m * BATCH + b) * DIM + n;  // m = s, n = d
        hr[idx] = f2bf(x * bf2f(r[idx]));
    );
}

// h2 = silu(hr @ w_h^T + b_h) ; out = query + u*(h2 - query)
__global__ __launch_bounds__(256) void gemm_h_kernel(const unsigned short* __restrict__ hr,
                                                     const unsigned short* __restrict__ wh,
                                                     const float* __restrict__ bh,
                                                     const float* __restrict__ u,
                                                     const float* __restrict__ query,
                                                     float* __restrict__ out) {
    WAVE_TILE_SETUP();
    wmma_gemm<4, 4>(hr, DIM, wh, DIM, DIM, m0w, n0w, acc);
    EPILOGUE_LOOP(
        x += bh[n];
        float h2 = x * sigmoidf_(x);
        size_t idx = (size_t)m * DIM + n;
        float qv = query[idx];
        out[idx] = qv + u[idx] * (h2 - qv);
    );
}

// ---------- launcher ----------
extern "C" void kernel_launch(void* const* d_in, const int* in_sizes, int n_in,
                              void* d_out, int out_size, void* d_ws, size_t ws_size,
                              hipStream_t stream) {
    const float* query = (const float*)d_in[0];
    const float* key   = (const float*)d_in[1];
    const float* w_q   = (const float*)d_in[2];
    const float* b_q   = (const float*)d_in[3];
    const float* w_k   = (const float*)d_in[4];
    const float* b_k   = (const float*)d_in[5];
    const float* w_v   = (const float*)d_in[6];
    const float* b_v   = (const float*)d_in[7];
    const float* w_h   = (const float*)d_in[8];
    const float* b_h   = (const float*)d_in[9];
    const float* ln_g  = (const float*)d_in[10];
    const float* ln_b  = (const float*)d_in[11];
    const float* rel   = (const float*)d_in[12];
    float* out = (float*)d_out;
    (void)in_sizes; (void)n_in; (void)out_size; (void)ws_size;

    char* base = (char*)d_ws;
    size_t off = 0;
    auto alloc = [&](size_t bytes) -> void* {
        void* p = base + off;
        off = (off + bytes + 255) & ~(size_t)255;
        return p;
    };
    unsigned short* qn   = (unsigned short*)alloc((size_t)MROWS * DIM * 2);
    unsigned short* keyb = (unsigned short*)alloc((size_t)MROWS * DIM * 2);
    unsigned short* wqb  = (unsigned short*)alloc((size_t)(2 * DIM + ZDIM) * DIM * 2);
    unsigned short* wkb  = (unsigned short*)alloc((size_t)ZDIM * DIM * 2);
    unsigned short* wvb  = (unsigned short*)alloc((size_t)DIM * DIM * 2);
    unsigned short* whb  = (unsigned short*)alloc((size_t)DIM * DIM * 2);
    float*          u    = (float*)alloc((size_t)MROWS * DIM * 4);
    unsigned short* rbuf = (unsigned short*)alloc((size_t)MROWS * DIM * 2);
    unsigned short* qbuf = (unsigned short*)alloc((size_t)MROWS * ZDIM * 2);
    unsigned short* kbuf = (unsigned short*)alloc((size_t)MROWS * ZDIM * 2);
    unsigned short* vt   = (unsigned short*)alloc((size_t)BATCH * DIM * S_LEN * 2);
    unsigned short* attn = (unsigned short*)alloc((size_t)BATCH * S_LEN * S_LEN * 2);
    unsigned short* hr   = (unsigned short*)alloc((size_t)MROWS * DIM * 2);

    auto cast = [&](const float* in, unsigned short* o, size_t n) {
        cast_kernel<<<dim3((unsigned)((n + 255) / 256)), dim3(256), 0, stream>>>(in, o, (int)n);
    };
    cast(key, keyb, (size_t)MROWS * DIM);
    cast(w_q, wqb, (size_t)(2 * DIM + ZDIM) * DIM);
    cast(w_k, wkb, (size_t)ZDIM * DIM);
    cast(w_v, wvb, (size_t)DIM * DIM);
    cast(w_h, whb, (size_t)DIM * DIM);

    ln_kernel<<<dim3(MROWS), dim3(256), 0, stream>>>(query, ln_g, ln_b, qn);

    gemm_q_kernel<<<dim3(MROWS / BLK_M, (2 * DIM + ZDIM) / BLK_N), dim3(256), 0, stream>>>(
        qn, wqb, b_q, u, rbuf, qbuf);
    gemm_k_kernel<<<dim3(MROWS / BLK_M, 1), dim3(256), 0, stream>>>(keyb, wkb, b_k, kbuf);
    gemm_v_kernel<<<dim3(MROWS / BLK_M, DIM / BLK_N), dim3(256), 0, stream>>>(keyb, wvb, b_v, vt);

    qk_kernel<<<dim3(S_LEN / BLK_M, S_LEN / BLK_N, BATCH), dim3(256), 0, stream>>>(
        qbuf, kbuf, rel, attn);
    av_kernel<<<dim3(S_LEN / BLK_M, DIM / BLK_N, BATCH), dim3(256), 0, stream>>>(
        attn, vt, rbuf, hr);

    gemm_h_kernel<<<dim3(MROWS / BLK_M, DIM / BLK_N), dim3(256), 0, stream>>>(
        hr, whb, b_h, u, query, out);
}
